// BiLSTM_CNN_21534966022929
// MI455X (gfx1250) — matrix operand order repaired and checked by reading, not compiled, not run on hardware
//
#include <hip/hip_runtime.h>
#include <hip/hip_bf16.h>

// ---------------------------------------------------------------------------
// CDNA5 (gfx1250) implementation: wave32, bf16 WMMA 16x16x32 for all GEMMs.
// All WMMA operands are staged in "fragment order" so each lane loads its
// 32-byte fragment contiguously (2x b128) instead of 16 scalar u16 loads.
// ---------------------------------------------------------------------------

typedef __attribute__((ext_vector_type(16))) __bf16 v16bf;
typedef __attribute__((ext_vector_type(8)))  float  v8f;

#define EPSV 1e-3f

__device__ inline v8f zero8() {
  v8f z;
#pragma unroll
  for (int i = 0; i < 8; ++i) z[i] = 0.f;
  return z;
}

__device__ inline v8f wmma_bf16(v16bf a, v16bf b, v8f c) {
  return __builtin_amdgcn_wmma_f32_16x16x32_bf16(false, a, false, b, (short)0, c,
                                                 false, false);
}

// ---- fragment-order slot mapping (16x32 bf16 tile -> 32 lanes x 16 elems) ----
// A tile (MxK=16x32): lane = m + 16*((k>>3)&1); within lane, elems 0..7 hold
// k in [kb,kb+8), elems 8..15 hold k in [kb+16,kb+24), kb = 8*((k>>3)&1).
__device__ inline int a_slot(int m, int k) {       // m in [0,16), k in [0,32)
  int hi = (k >> 3) & 1;
  int kp = k - (hi << 3);
  int elem = (kp < 8) ? kp : (kp - 8);
  return ((m + (hi << 4)) << 4) + elem;
}
// B tile (KxN=32x16): lane = n + 16*((k>>3)&1); elem = (k & 7) + 8*(k>=16).
__device__ inline int b_slot(int k, int n) {       // k in [0,32), n in [0,16)
  return ((n + (((k >> 3) & 1) << 4)) << 4) + (k & 7) + ((k >> 4) << 3);
}

__device__ inline v16bf frag16(const __bf16* base) {  // lane's 32B fragment
  return *(const v16bf*)(base + (threadIdx.x & 31) * 16);
}

// ---------------------------------------------------------------------------
// fp32 -> bf16 conversion (row-major)
// ---------------------------------------------------------------------------
__global__ void cvt_kernel(const float* __restrict__ s, __bf16* __restrict__ d, int n) {
  int i = blockIdx.x * blockDim.x + threadIdx.x;
  if (i < n) d[i] = (__bf16)s[i];
}

// fp32 R[U][4U] -> fragment-ordered bf16: tile nt (n/16), kblock (k/32),
// slot layout ((nt*(U/32)+kb)*512 + b_slot(k&31, n&15))
__global__ void swizzle_r_kernel(const float* __restrict__ R, __bf16* __restrict__ out,
                                 int U) {
  int fourU = 4 * U;
  int idx = blockIdx.x * blockDim.x + threadIdx.x;
  if (idx >= U * fourU) return;
  int k = idx / fourU, n = idx % fourU;
  int nt = n >> 4, kb = k >> 5;
  int slot = ((nt * (U >> 5) + kb) << 9) + b_slot(k & 31, n & 15);
  out[slot] = (__bf16)R[idx];
}

// ---------------------------------------------------------------------------
// Conv1D(K) + bias + ReLU + BN + MaxPool(2)  -> bf16 slice of [512,128,192]
// grid: (512, 8) ; block: 256
// ---------------------------------------------------------------------------
__global__ __launch_bounds__(256)
void conv_block_kernel(const float* __restrict__ x, const float* __restrict__ w,
                       const float* __restrict__ bias,
                       const float* __restrict__ gg, const float* __restrict__ bb,
                       const float* __restrict__ mm, const float* __restrict__ vv,
                       __bf16* __restrict__ out, int KW, int coff) {
  __shared__ float xs[40][78];
  int b = blockIdx.x, tile = blockIdx.y, tid = threadIdx.x;
  int pad = KW >> 1;
  int t0 = tile * 32 - pad;
  int rows = 32 + KW - 1;
  for (int e = tid; e < rows * 78; e += 256) {
    int r = e / 78, c = e % 78;
    int t = t0 + r;
    xs[r][c] = (t >= 0 && t < 256) ? x[((size_t)b * 256 + t) * 78 + c] : 0.f;
  }
  __syncthreads();
#pragma unroll
  for (int i = 0; i < 4; ++i) {
    int oi = tid + i * 256;
    int tl = oi >> 6;
    int c  = oi & 63;
    float scale = gg[c] * rsqrtf(vv[c] + EPSV);
    float shift = bb[c] - mm[c] * scale;
    float best = -3.0e38f;
    for (int p = 0; p < 2; ++p) {
      float acc = bias[c];
      int rbase = tl * 2 + p;
      for (int k = 0; k < KW; ++k)
        for (int ci = 0; ci < 78; ++ci)
          acc += xs[rbase + k][ci] * w[((size_t)k * 78 + ci) * 64 + c];
      float val = fmaxf(acc, 0.f) * scale + shift;
      best = fmaxf(best, val);
    }
    out[((size_t)b * 128 + tile * 16 + tl) * 192 + coff + c] = (__bf16)best;
  }
}

// ---------------------------------------------------------------------------
// Generic GEMM: out = act(A[M,K] @ W[K,N] + bias), bf16 in / f32 acc.
// block tile 64x64, 8 waves, LDS tiles stored in fragment order.
// ---------------------------------------------------------------------------
template <bool HB, int ACT, bool OF, bool OB>
__global__ __launch_bounds__(256)
void gemm_bf16_kernel(const __bf16* __restrict__ A, const __bf16* __restrict__ W,
                      const float* __restrict__ bias,
                      float* __restrict__ outF, __bf16* __restrict__ outB,
                      int M, int K, int N) {
  __shared__ alignas(32) __bf16 As[4 * 512];  // 4 M-subtiles, fragment order
  __shared__ alignas(32) __bf16 Bs[4 * 512];  // 4 N-subtiles, fragment order
  int tid = threadIdx.x;
  int m0 = blockIdx.x * 64;
  int n0 = blockIdx.y * 64;
  int wave = tid >> 5;
  int lane = tid & 31;
  int wm = wave & 3;
  int wn = wave >> 2;
  v8f acc0 = zero8(), acc1 = zero8();
  for (int kk = 0; kk < K; kk += 32) {
    for (int idx = tid; idx < 2048; idx += 256) {
      int m = idx >> 5, kc = idx & 31;
      As[((m >> 4) << 9) + a_slot(m & 15, kc)] = A[(size_t)(m0 + m) * K + kk + kc];
    }
    for (int idx = tid; idx < 2048; idx += 256) {
      int k = idx >> 6, nc = idx & 63;
      Bs[((nc >> 4) << 9) + b_slot(k, nc & 15)] = W[(size_t)(kk + k) * N + n0 + nc];
    }
    __syncthreads();
    v16bf a  = frag16(&As[wm << 9]);
    v16bf b0 = frag16(&Bs[(wn * 2 + 0) << 9]);
    v16bf b1 = frag16(&Bs[(wn * 2 + 1) << 9]);
    acc0 = wmma_bf16(a, b0, acc0);
    acc1 = wmma_bf16(a, b1, acc1);
    __syncthreads();
  }
  int hi = (lane & 16) ? 8 : 0;
  int nc = lane & 15;
  int na = n0 + wn * 32 + nc;
  int nb = na + 16;
  float bna = HB ? bias[na] : 0.f;
  float bnb = HB ? bias[nb] : 0.f;
#pragma unroll
  for (int r = 0; r < 8; ++r) {
    int m = m0 + wm * 16 + r + hi;
    float v0 = acc0[r] + bna;
    float v1 = acc1[r] + bnb;
    if (ACT) { v0 = fmaxf(v0, 0.f); v1 = fmaxf(v1, 0.f); }
    if (OF) { outF[(size_t)m * N + na] = v0; outF[(size_t)m * N + nb] = v1; }
    if (OB) { outB[(size_t)m * N + na] = (__bf16)v0; outB[(size_t)m * N + nb] = (__bf16)v1; }
  }
}

// ---------------------------------------------------------------------------
// Temporal mean
// ---------------------------------------------------------------------------
__global__ void mean_t_kernel(const float* __restrict__ src, float* __restrict__ dstF,
                              __bf16* __restrict__ dstB, int Bn, int Tn, int C) {
  int i = blockIdx.x * blockDim.x + threadIdx.x;
  if (i >= Bn * C) return;
  int b = i / C, c = i % C;
  float s = 0.f;
  for (int t = 0; t < Tn; ++t) s += src[((size_t)b * Tn + t) * C + c];
  s /= (float)Tn;
  if (dstF) dstF[i] = s;
  if (dstB) dstB[i] = (__bf16)s;
}

__global__ __launch_bounds__(128)
void se_excite_kernel(const float* __restrict__ s, const float* __restrict__ w1,
                      const float* __restrict__ b1, const float* __restrict__ w2,
                      const float* __restrict__ b2, float* __restrict__ e) {
  __shared__ float hid[8];
  int b = blockIdx.x, tid = threadIdx.x;
  if (tid < 8) {
    float a = b1[tid];
    for (int c = 0; c < 128; ++c) a += s[(size_t)b * 128 + c] * w1[c * 8 + tid];
    hid[tid] = fmaxf(a, 0.f);
  }
  __syncthreads();
  float a = b2[tid];
  for (int j = 0; j < 8; ++j) a += hid[j] * w2[j * 128 + tid];
  e[(size_t)b * 128 + tid] = 1.f / (1.f + __expf(-a));
}

__global__ void se_scale_kernel(const float* __restrict__ fused,
                                const float* __restrict__ e, __bf16* __restrict__ out) {
  size_t i = (size_t)blockIdx.x * blockDim.x + threadIdx.x;
  if (i >= (size_t)512 * 128 * 128) return;
  int c = (int)(i & 127);
  size_t b = i >> 14;
  out[i] = (__bf16)(fused[i] * e[b * 128 + c]);
}

__global__ void bn_kernel(const float* __restrict__ x, const float* __restrict__ g,
                          const float* __restrict__ bb, const float* __restrict__ m,
                          const float* __restrict__ v, float* __restrict__ outF,
                          __bf16* __restrict__ outB, size_t n, int C) {
  size_t i = (size_t)blockIdx.x * blockDim.x + threadIdx.x;
  if (i >= n) return;
  int c = (int)(i % C);
  float y = (x[i] - m[c]) * rsqrtf(v[c] + EPSV) * g[c] + bb[c];
  if (outF) outF[i] = y;
  if (outB) outB[i] = (__bf16)y;
}

// ---------------------------------------------------------------------------
// LSTM recurrence: one block = 16 batch rows; h/c/z in LDS; h stored in
// fragment order; R pre-swizzled in global fragment order.
// ---------------------------------------------------------------------------
__global__ __launch_bounds__(256)
void lstm_kernel(const float* __restrict__ xp, const __bf16* __restrict__ R,
                 float* __restrict__ out, int T, int U, int ostride, int rev) {
  __shared__ alignas(32) __bf16 hfrag[4 * 512];  // (U/32) k-blocks, frag order
  __shared__ float cst[16][128];
  __shared__ float zz[16][512];
  const int fourU = 4 * U;
  const int nkb = U >> 5;
  int tid = threadIdx.x;
  int wave = tid >> 5, lane = tid & 31;
  int b0 = blockIdx.x * 16;
  int ntiles = fourU >> 4;
  int perWave = ntiles >> 3;
  for (int e = tid; e < nkb * 512; e += 256) hfrag[e] = (__bf16)0.f;
  for (int e = tid; e < 16 * U; e += 256) cst[e / U][e % U] = 0.f;
  __syncthreads();
  for (int s = 0; s < T; ++s) {
    int t = rev ? (T - 1 - s) : s;
    // z = h @ R
    for (int i = 0; i < perWave; ++i) {
      int nt = wave * perWave + i;
      v8f acc = zero8();
      for (int kb = 0; kb < nkb; ++kb) {
        v16bf a = frag16(&hfrag[kb << 9]);
        v16bf b = *(const v16bf*)(R + (((size_t)(nt * nkb + kb)) << 9) + lane * 16);
        acc = wmma_bf16(a, b, acc);
      }
      int hi = (lane & 16) ? 8 : 0, nc = lane & 15;
#pragma unroll
      for (int r = 0; r < 8; ++r) zz[r + hi][nt * 16 + nc] = acc[r];
    }
    __syncthreads();
    // gates
    for (int e = tid; e < 16 * U; e += 256) {
      int m = e / U, u = e % U;
      const float* xr = xp + ((size_t)(b0 + m) * T + t) * fourU;
      float zi = zz[m][u]         + xr[u];
      float zf = zz[m][U + u]     + xr[U + u];
      float zg = zz[m][2 * U + u] + xr[2 * U + u];
      float zo = zz[m][3 * U + u] + xr[3 * U + u];
      float si = 1.f / (1.f + __expf(-zi));
      float sf = 1.f / (1.f + __expf(-zf));
      float so = 1.f / (1.f + __expf(-zo));
      float c2 = sf * cst[m][u] + si * tanhf(zg);
      float h2 = so * tanhf(c2);
      cst[m][u] = c2;
      hfrag[((u >> 5) << 9) + a_slot(m, u & 31)] = (__bf16)h2;
      out[((size_t)(b0 + m) * T + t) * ostride + u] = h2;
    }
    __syncthreads();
  }
}

// ---------------------------------------------------------------------------
// MHSA per (batch, head, 64-query tile). Q/K staged swizzled in a union
// region that is later reused for fragment-ordered probabilities.
// ---------------------------------------------------------------------------
__global__ __launch_bounds__(256)
void attn_kernel(const __bf16* __restrict__ q, const __bf16* __restrict__ k,
                 const __bf16* __restrict__ v, __bf16* __restrict__ outB,
                 int T, int dm, int H) {
  __shared__ float scratch[64 * 128];                // logits (32KB)
  __shared__ alignas(32) __bf16 vsw[8 * 512];        // V frag order (8KB)
  __shared__ alignas(32) __bf16 ureg[16 * 512];      // q(4)+k(8) then probs(16)
  const int dh = dm / H;  // 32
  int bh = blockIdx.x;
  int b = bh / H, h = bh % H;
  int q0 = blockIdx.y * 64;
  int tid = threadIdx.x, wave = tid >> 5, lane = tid & 31;
  const __bf16* qb = q + (size_t)b * T * dm + h * dh;
  const __bf16* kp = k + (size_t)b * T * dm + h * dh;
  const __bf16* vp = v + (size_t)b * T * dm + h * dh;
  __bf16* qsw = ureg;              // 4 tiles
  __bf16* ksw = ureg + 4 * 512;    // 8 tiles
  __bf16* absw = ureg;             // 16 tiles (reused after logits)
  // stage V fragments: B tile (nt over dh halves, kb over 4 time blocks)
  for (int idx = tid; idx < 4096; idx += 256) {
    int r = idx >> 5, c = idx & 31;
    vsw[((((c >> 4) << 2) + (r >> 5)) << 9) + b_slot(r & 31, c & 15)] = vp[(size_t)r * dm + c];
  }
  // stage Q (A-frags) and K (B^T-frags)
  for (int idx = tid; idx < 2048; idx += 256) {
    int m = idx >> 5, d = idx & 31;
    qsw[((m >> 4) << 9) + a_slot(m & 15, d)] = qb[(size_t)(q0 + m) * dm + d];
  }
  for (int idx = tid; idx < 4096; idx += 256) {
    int key = idx >> 5, d = idx & 31;
    ksw[((key >> 4) << 9) + b_slot(d, key & 15)] = kp[(size_t)key * dm + d];
  }
  __syncthreads();
  // logits: 4x8 tiles, K=32 single WMMA each
  for (int i = 0; i < 4; ++i) {
    int tile = wave * 4 + i;
    int mt = tile & 3, nt = tile >> 2;
    v16bf a  = frag16(&qsw[mt << 9]);
    v16bf bt = frag16(&ksw[nt << 9]);
    v8f acc = wmma_bf16(a, bt, zero8());
    int hi = (lane & 16) ? 8 : 0, nc = lane & 15;
#pragma unroll
    for (int r = 0; r < 8; ++r)
      scratch[(mt * 16 + r + hi) * 128 + nt * 16 + nc] = acc[r] * 0.1767766953f;
  }
  __syncthreads();
  if (tid < 64) {
    float* row = &scratch[tid * 128];
    float mx = -3.0e38f;
    for (int j = 0; j < 128; ++j) mx = fmaxf(mx, row[j]);
    float sum = 0.f;
    for (int j = 0; j < 128; ++j) { float p = __expf(row[j] - mx); row[j] = p; sum += p; }
    float inv = 1.f / sum;
    int mt = tid >> 4, mrow = tid & 15;
    for (int j = 0; j < 128; ++j)
      absw[(((mt << 2) + (j >> 5)) << 9) + a_slot(mrow, j & 31)] = (__bf16)(row[j] * inv);
  }
  __syncthreads();
  // out = P @ V : 4x2 tiles, K loop 4
  {
    int mt = wave & 3, nt = wave >> 2;
    v8f acc = zero8();
    for (int kb = 0; kb < 4; ++kb) {
      v16bf a = frag16(&absw[((mt << 2) + kb) << 9]);
      v16bf bv = frag16(&vsw[((nt << 2) + kb) << 9]);
      acc = wmma_bf16(a, bv, acc);
    }
    int hi = (lane & 16) ? 8 : 0, nc = lane & 15;
#pragma unroll
    for (int r = 0; r < 8; ++r) {
      int m = q0 + mt * 16 + r + hi;
      outB[((size_t)b * T + m) * dm + h * dh + nt * 16 + nc] = (__bf16)acc[r];
    }
  }
}

__global__ void d3_kernel(const float* __restrict__ d2, const float* __restrict__ w,
                          const float* __restrict__ b, float* __restrict__ out) {
  int i = blockIdx.x * blockDim.x + threadIdx.x;
  if (i >= 512) return;
  float acc = b[0];
  for (int j = 0; j < 128; ++j) acc += d2[(size_t)i * 128 + j] * w[j];
  out[i] = 1.f / (1.f + __expf(-acc));
}

// ---------------------------------------------------------------------------
// Host orchestration
// ---------------------------------------------------------------------------
static void gemm(const __bf16* A, const __bf16* W, const float* bias, float* outF,
                 __bf16* outB, int M, int K, int N, int act, hipStream_t s) {
  dim3 g(M / 64, N / 64);
  if (outF && !outB) {
    if (act) gemm_bf16_kernel<true, 1, true, false><<<g, 256, 0, s>>>(A, W, bias, outF, outB, M, K, N);
    else     gemm_bf16_kernel<true, 0, true, false><<<g, 256, 0, s>>>(A, W, bias, outF, outB, M, K, N);
  } else if (!outF && outB) {
    gemm_bf16_kernel<true, 0, false, true><<<g, 256, 0, s>>>(A, W, bias, outF, outB, M, K, N);
  } else {
    gemm_bf16_kernel<true, 0, true, true><<<g, 256, 0, s>>>(A, W, bias, outF, outB, M, K, N);
  }
}

extern "C" void kernel_launch(void* const* d_in, const int* in_sizes, int n_in,
                              void* d_out, int out_size, void* d_ws, size_t ws_size,
                              hipStream_t stream) {
  (void)in_sizes; (void)n_in; (void)out_size; (void)ws_size;
  const float* X      = (const float*)d_in[0];
  const float* c5w    = (const float*)d_in[1];
  const float* c7w    = (const float*)d_in[2];
  const float* c9w    = (const float*)d_in[3];
  const float* fusew  = (const float*)d_in[4];
  const float* sew1   = (const float*)d_in[5];
  const float* sew2   = (const float*)d_in[6];
  const float* l1fk   = (const float*)d_in[7];
  const float* l1fr   = (const float*)d_in[8];
  const float* l1bk   = (const float*)d_in[9];
  const float* l1br   = (const float*)d_in[10];
  const float* l2fk   = (const float*)d_in[11];
  const float* l2fr   = (const float*)d_in[12];
  const float* l2bk   = (const float*)d_in[13];
  const float* l2br   = (const float*)d_in[14];
  const float* wq     = (const float*)d_in[15];
  const float* wk     = (const float*)d_in[16];
  const float* wv     = (const float*)d_in[17];
  const float* wo     = (const float*)d_in[18];
  const float* d1w    = (const float*)d_in[19];
  const float* d2w    = (const float*)d_in[20];
  const float* d3w    = (const float*)d_in[21];
  const float* c5b    = (const float*)d_in[22];
  const float* c7b    = (const float*)d_in[23];
  const float* c9b    = (const float*)d_in[24];
  const float* fuseb  = (const float*)d_in[25];
  const float* seb1   = (const float*)d_in[26];
  const float* seb2   = (const float*)d_in[27];
  const float* l1fb   = (const float*)d_in[28];
  const float* l1bb   = (const float*)d_in[29];
  const float* l2fb   = (const float*)d_in[30];
  const float* l2bb   = (const float*)d_in[31];
  const float* bq     = (const float*)d_in[32];
  const float* bk     = (const float*)d_in[33];
  const float* bv     = (const float*)d_in[34];
  const float* bo     = (const float*)d_in[35];
  const float* d1b    = (const float*)d_in[36];
  const float* d2b    = (const float*)d_in[37];
  const float* d3b    = (const float*)d_in[38];
  const float* bn5[4]  = {(const float*)d_in[39], (const float*)d_in[40], (const float*)d_in[41], (const float*)d_in[42]};
  const float* bn7[4]  = {(const float*)d_in[43], (const float*)d_in[44], (const float*)d_in[45], (const float*)d_in[46]};
  const float* bn9[4]  = {(const float*)d_in[47], (const float*)d_in[48], (const float*)d_in[49], (const float*)d_in[50]};
  const float* bnl1[4] = {(const float*)d_in[51], (const float*)d_in[52], (const float*)d_in[53], (const float*)d_in[54]};
  const float* bnl2[4] = {(const float*)d_in[55], (const float*)d_in[56], (const float*)d_in[57], (const float*)d_in[58]};
  const float* bnd1[4] = {(const float*)d_in[59], (const float*)d_in[60], (const float*)d_in[61], (const float*)d_in[62]};
  const float* bnd2[4] = {(const float*)d_in[63], (const float*)d_in[64], (const float*)d_in[65], (const float*)d_in[66]};

  char* ws = (char*)d_ws;
  const size_t OFF_FUSEDIN = 0;                  // bf16 [512,128,192]
  const size_t OFF_FUSED   = 25165824;           // f32  [512,128,128]
  const size_t OFF_SMEAN   = 58720256;           // f32  [512,128]
  const size_t OFF_E       = 58982400;           // f32  [512,128]
  const size_t OFF_HSEB    = 59244544;           // bf16 [512,128,128]
  const size_t OFF_WBF     = 76021760;           // bf16 weight pool
  const size_t OFF_XP1F    = 78118912;           // f32  [512,128,512]
  const size_t OFF_XP1B    = 212336640;          // f32  [512,128,512]
  const size_t OFF_H1      = 346554368;          // f32  [512,128,256]
  const size_t OFF_H1B     = 413663232;          // bf16 [512,128,256]
  const size_t OFF_XP2F    = OFF_XP1F;
  const size_t OFF_XP2B    = OFF_XP1B;
  const size_t OFF_H2      = OFF_FUSED;
  const size_t OFF_H2B     = OFF_HSEB;
  const size_t OFF_QB      = OFF_XP1F;
  const size_t OFF_KB      = OFF_XP1F + 16777216;
  const size_t OFF_VB      = OFF_XP1F + 33554432;
  const size_t OFF_ATTNB   = OFF_XP1B;
  const size_t OFF_PROJ    = OFF_H1;
  const size_t OFF_GMEANF  = OFF_H1B;
  const size_t OFF_GMEANB  = OFF_H1B + 524288;
  const size_t OFF_D1F     = OFF_H1B + 786432;
  const size_t OFF_D1B     = OFF_H1B + 1310720;
  const size_t OFF_D2F     = OFF_H1B + 1572864;
  const size_t OFF_D2B     = OFF_H1B + 1835008;

  __bf16* wpool = (__bf16*)(ws + OFF_WBF);
  __bf16* fusewB = wpool + 0;
  __bf16* l1fkB  = wpool + 24576;
  __bf16* l1bkB  = wpool + 90112;
  __bf16* l1frB  = wpool + 155648;   // fragment-ordered
  __bf16* l1brB  = wpool + 221184;   // fragment-ordered
  __bf16* l2fkB  = wpool + 286720;
  __bf16* l2bkB  = wpool + 352256;
  __bf16* l2frB  = wpool + 417792;   // fragment-ordered
  __bf16* l2brB  = wpool + 434176;   // fragment-ordered
  __bf16* wqB    = wpool + 450560;
  __bf16* wkB    = wpool + 483328;
  __bf16* wvB    = wpool + 516096;
  __bf16* woB    = wpool + 548864;
  __bf16* d1wB   = wpool + 614400;
  __bf16* d2wB   = wpool + 679936;

  auto cvt = [&](const float* s, __bf16* d, int n) {
    cvt_kernel<<<(n + 255) / 256, 256, 0, stream>>>(s, d, n);
  };
  cvt(fusew, fusewB, 24576);
  cvt(l1fk, l1fkB, 65536);  cvt(l1bk, l1bkB, 65536);
  cvt(l2fk, l2fkB, 65536);  cvt(l2bk, l2bkB, 65536);
  cvt(wq, wqB, 32768);      cvt(wk, wkB, 32768);      cvt(wv, wvB, 32768);
  cvt(wo, woB, 65536);      cvt(d1w, d1wB, 65536);    cvt(d2w, d2wB, 32768);
  swizzle_r_kernel<<<256, 256, 0, stream>>>(l1fr, l1frB, 128);
  swizzle_r_kernel<<<256, 256, 0, stream>>>(l1br, l1brB, 128);
  swizzle_r_kernel<<<64, 256, 0, stream>>>(l2fr, l2frB, 64);
  swizzle_r_kernel<<<64, 256, 0, stream>>>(l2br, l2brB, 64);

  __bf16* fusedIn = (__bf16*)(ws + OFF_FUSEDIN);
  conv_block_kernel<<<dim3(512, 8), 256, 0, stream>>>(X, c5w, c5b, bn5[0], bn5[1], bn5[2], bn5[3], fusedIn, 5, 0);
  conv_block_kernel<<<dim3(512, 8), 256, 0, stream>>>(X, c7w, c7b, bn7[0], bn7[1], bn7[2], bn7[3], fusedIn, 7, 64);
  conv_block_kernel<<<dim3(512, 8), 256, 0, stream>>>(X, c9w, c9b, bn9[0], bn9[1], bn9[2], bn9[3], fusedIn, 9, 128);

  const int M = 512 * 128;
  float* fused = (float*)(ws + OFF_FUSED);
  gemm(fusedIn, fusewB, fuseb, fused, nullptr, M, 192, 128, 0, stream);

  float* smean = (float*)(ws + OFF_SMEAN);
  float* evec  = (float*)(ws + OFF_E);
  mean_t_kernel<<<(512 * 128 + 255) / 256, 256, 0, stream>>>(fused, smean, nullptr, 512, 128, 128);
  se_excite_kernel<<<512, 128, 0, stream>>>(smean, sew1, seb1, sew2, seb2, evec);
  __bf16* hseB = (__bf16*)(ws + OFF_HSEB);
  se_scale_kernel<<<(512 * 128 * 128 + 255) / 256, 256, 0, stream>>>(fused, evec, hseB);

  float* xp1f = (float*)(ws + OFF_XP1F);
  float* xp1b = (float*)(ws + OFF_XP1B);
  gemm(hseB, l1fkB, l1fb, xp1f, nullptr, M, 128, 512, 0, stream);
  gemm(hseB, l1bkB, l1bb, xp1b, nullptr, M, 128, 512, 0, stream);
  float* h1 = (float*)(ws + OFF_H1);
  lstm_kernel<<<32, 256, 0, stream>>>(xp1f, l1frB, h1, 128, 128, 256, 0);
  lstm_kernel<<<32, 256, 0, stream>>>(xp1b, l1brB, h1 + 128, 128, 128, 256, 1);
  __bf16* h1B = (__bf16*)(ws + OFF_H1B);
  {
    size_t n = (size_t)512 * 128 * 256;
    bn_kernel<<<(unsigned)((n + 255) / 256), 256, 0, stream>>>(h1, bnl1[0], bnl1[1], bnl1[2], bnl1[3], h1, h1B, n, 256);
  }

  float* xp2f = (float*)(ws + OFF_XP2F);
  float* xp2b = (float*)(ws + OFF_XP2B);
  gemm(h1B, l2fkB, l2fb, xp2f, nullptr, M, 256, 256, 0, stream);
  gemm(h1B, l2bkB, l2bb, xp2b, nullptr, M, 256, 256, 0, stream);
  float* h2 = (float*)(ws + OFF_H2);
  lstm_kernel<<<32, 256, 0, stream>>>(xp2f, l2frB, h2, 128, 64, 128, 0);
  lstm_kernel<<<32, 256, 0, stream>>>(xp2b, l2brB, h2 + 64, 128, 64, 128, 1);
  __bf16* h2B = (__bf16*)(ws + OFF_H2B);
  {
    size_t n = (size_t)512 * 128 * 128;
    bn_kernel<<<(unsigned)((n + 255) / 256), 256, 0, stream>>>(h2, bnl2[0], bnl2[1], bnl2[2], bnl2[3], h2, h2B, n, 128);
  }

  __bf16* qB = (__bf16*)(ws + OFF_QB);
  __bf16* kB = (__bf16*)(ws + OFF_KB);
  __bf16* vB = (__bf16*)(ws + OFF_VB);
  gemm(h2B, wqB, bq, nullptr, qB, M, 128, 256, 0, stream);
  gemm(h2B, wkB, bk, nullptr, kB, M, 128, 256, 0, stream);
  gemm(h2B, wvB, bv, nullptr, vB, M, 128, 256, 0, stream);
  __bf16* attnB = (__bf16*)(ws + OFF_ATTNB);
  attn_kernel<<<dim3(512 * 8, 2), 256, 0, stream>>>(qB, kB, vB, attnB, 128, 256, 8);
  float* projF = (float*)(ws + OFF_PROJ);
  gemm(attnB, woB, bo, projF, nullptr, M, 256, 256, 0, stream);

  float*  gmeanF = (float*)(ws + OFF_GMEANF);
  __bf16* gmeanB = (__bf16*)(ws + OFF_GMEANB);
  mean_t_kernel<<<(512 * 256 + 255) / 256, 256, 0, stream>>>(projF, gmeanF, gmeanB, 512, 128, 256);

  float*  d1F = (float*)(ws + OFF_D1F);
  __bf16* d1B = (__bf16*)(ws + OFF_D1B);
  gemm(gmeanB, d1wB, d1b, d1F, nullptr, 512, 256, 256, 1, stream);
  bn_kernel<<<(512 * 256 + 255) / 256, 256, 0, stream>>>(d1F, bnd1[0], bnd1[1], bnd1[2], bnd1[3], d1F, d1B, (size_t)512 * 256, 256);

  float*  d2F = (float*)(ws + OFF_D2F);
  __bf16* d2B = (__bf16*)(ws + OFF_D2B);
  gemm(d1B, d2wB, d2b, d2F, nullptr, 512, 256, 128, 1, stream);
  bn_kernel<<<(512 * 128 + 255) / 256, 256, 0, stream>>>(d2F, bnd2[0], bnd2[1], bnd2[2], bnd2[3], d2F, d2B, (size_t)512 * 128, 128);

  d3_kernel<<<2, 256, 0, stream>>>(d2F, d3w, d3b, (float*)d_out);
}
